// InstConv2dv3_44246753083743
// MI455X (gfx1250) — compile-verified, matrix operand order checked by reading, not run on hardware
//
#include <hip/hip_runtime.h>
#include <hip/hip_bf16.h>
#include <stdint.h>

// ---------------------------------------------------------------------------
// InstConv2dv3: 3x3 conv with cumprod-gated weights, implicit-GEMM via
// v_wmma_f32_16x16x32_bf16 with 2-term bf16 split (hi+lo) for ~fp32 accuracy.
// B=16, CIN=COUT=256, H=W=64, K=3, PAD=1.
//
// v3b: A-tile staging via GLOBAL_LOAD_ASYNC_TO_LDS_B128 (ASYNCcnt),
//      overlapped with the B-panel fp32->bf16 conversion. The builtin takes
//      (global int4-vec*, lds int4-vec*, imm offset, imm cpol).
// ---------------------------------------------------------------------------

#define Bn   16
#define CIN  256
#define COUT 256
#define Hn   64
#define Wn   64
#define HW   (Hn * Wn)

#if defined(__has_builtin)
# if __has_builtin(__builtin_amdgcn_global_load_async_to_lds_b128) && \
     __has_builtin(__builtin_amdgcn_s_wait_asynccnt)
#  define USE_ASYNC_A 1
# endif
#endif

typedef __attribute__((ext_vector_type(16))) __bf16 v16bf;
typedef __attribute__((ext_vector_type(8)))  float  v8f;

union FragU {               // 16 bf16 = 32 bytes = 2 x uint4
    uint4 q[2];
    v16bf v;
};

__device__ __forceinline__ unsigned short bf_bits(__bf16 b) {
    union { __bf16 b; unsigned short u; } c; c.b = b; return c.u;
}

#ifdef USE_ASYNC_A
typedef int vint4_t __attribute__((vector_size(16)));
typedef __attribute__((address_space(1))) vint4_t* glb_v4i;
typedef __attribute__((address_space(3))) vint4_t* lds_v4i;

// Global pointer: generic address == global address for the global aperture.
__device__ __forceinline__ glb_v4i to_glb(const void* p) {
    return (glb_v4i)(uintptr_t)p;
}
// LDS addrspace(3) pointer = low 32 bits of the generic address (ISA 10.2:
// LDS aperture addresses truncate to LDS_ADDR[31:0]).
__device__ __forceinline__ lds_v4i to_lds(const void* p) {
    return (lds_v4i)(uint32_t)(uintptr_t)p;
}
#endif

// ---------------------------------------------------------------------------
// K1: cumulative product of relu(gate) along cout (row 0 -> scale 1).
// ---------------------------------------------------------------------------
__global__ void scale_kernel(const float* __restrict__ gate,
                             float* __restrict__ scale) {
    const int ci = threadIdx.x;           // 0..255
    float p = 1.0f;
    for (int co = 0; co < COUT; ++co) {
        if (co > 0) {
            float g = gate[co * CIN + ci];
            p *= (g > 0.0f ? g : 0.0f);
        }
        scale[co * CIN + ci] = p;
    }
}

// ---------------------------------------------------------------------------
// K2: w_eff = conv_w * scale, split into bf16 hi/lo planes, layout [t][co][ci].
// ---------------------------------------------------------------------------
__global__ void wpack_kernel(const float* __restrict__ conv_w,
                             const float* __restrict__ scale,
                             unsigned short* __restrict__ whi,
                             unsigned short* __restrict__ wlo) {
    const int co = blockIdx.x;            // 0..255
    const int ci = threadIdx.x;           // 0..255
    const float s = scale[co * CIN + ci];
    #pragma unroll
    for (int t = 0; t < 9; ++t) {
        float w  = conv_w[(co * CIN + ci) * 9 + t] * s;
        __bf16 h = (__bf16)w;
        __bf16 l = (__bf16)(w - (float)h);
        whi[(size_t)t * (COUT * CIN) + co * CIN + ci] = bf_bits(h);
        wlo[(size_t)t * (COUT * CIN) + co * CIN + ci] = bf_bits(l);
    }
}

// ---------------------------------------------------------------------------
// K3: implicit-GEMM conv. Block = 64 couts x 64 w for one (n, h).
// 8 waves; wave tile = 16 couts x 32 w = two 16x16 WMMA accumulators.
// Staging loop: 3 dh rows x 8 chunks of 32 cin; each staging step feeds the
// three dw taps (18 WMMAs) from one padded B panel + three A tiles.
// ---------------------------------------------------------------------------
#define LSTR 40   // LDS row stride (halfwords): 16B-aligned runs, conflict-free

__global__ __launch_bounds__(256)
void conv_kernel(const float* __restrict__ in,
                 const unsigned short* __restrict__ whi,
                 const unsigned short* __restrict__ wlo,
                 const float* __restrict__ bias,
                 float* __restrict__ out) {
    // A: one 64x32 tile per tap of the current dh row, hi/lo planes.
    __shared__ __align__(16) unsigned short As_hi[3][64 * LSTR];
    __shared__ __align__(16) unsigned short As_lo[3][64 * LSTR];
    // B: padded panel, rows p = w_in + 1 for w_in in [-1, 64], cols = 32 cin.
    __shared__ __align__(16) unsigned short Bs_hi[66 * LSTR];
    __shared__ __align__(16) unsigned short Bs_lo[66 * LSTR];

    const int bx  = blockIdx.x;
    const int cb  = (bx & 3) * 64;        // cout tile base
    const int row = bx >> 2;              // 0..1023
    const int n   = row >> 6;
    const int h   = row & 63;

    const int tid  = threadIdx.x;
    const int wave = tid >> 5;
    const int lane = tid & 31;
    const int m    = lane & 15;
    const int kb   = (lane >= 16) ? 8 : 0;          // per-lane K run base
    const int co_sub = (wave & 3) * 16;             // wave's cout sub-tile
    const int w_sub  = (wave >> 2) * 32;            // wave's w sub-tile (2x16)

    // A fill mapping: thread -> (cout row, 8 contiguous cin)
    const int a_co = tid >> 2;            // 0..63
    const int a_cg = (tid & 3) * 8;
    // B fill mapping: thread -> (w column, 8 contiguous cin)
    const int b_w  = tid & 63;            // 0..63
    const int b_cg = (tid >> 6) * 8;      // 0,8,16,24

    // Pad rows p=0 (w=-1) and p=65 (w=64) are always zero; write them once.
    if (tid < 32) {
        Bs_hi[tid] = 0;             Bs_lo[tid] = 0;
        Bs_hi[65 * LSTR + tid] = 0; Bs_lo[65 * LSTR + tid] = 0;
    }

    v8f d0 = {};  // D tile (co_sub, w_sub)
    v8f d1 = {};  // D tile (co_sub, w_sub+16)

    #pragma unroll 1
    for (int th = 0; th < 3; ++th) {      // dh = th - 1
        const int hh = h + th - 1;
        const bool rv = (hh >= 0) && (hh < Hn);           // block-uniform
        const long long rowoff = ((long long)n * CIN * Hn + hh) * (long long)Wn;

        #pragma unroll 1
        for (int cc = 0; cc < CIN / 32; ++cc) {
            const int cib = cc * 32;
            __syncthreads();   // previous step's fragment reads are done

            // ---- stage A: three taps (dh, dw=-1/0/1), 64co x 32ci each.
            //      Async global->LDS copies overlap with the B conversion. --
            #pragma unroll
            for (int j = 0; j < 3; ++j) {
                const size_t g = (size_t)(th * 3 + j) * (COUT * CIN)
                               + (size_t)(cb + a_co) * CIN + cib + a_cg;
#ifdef USE_ASYNC_A
                __builtin_amdgcn_global_load_async_to_lds_b128(
                    to_glb(whi + g), to_lds(As_hi[j] + a_co * LSTR + a_cg), 0, 0);
                __builtin_amdgcn_global_load_async_to_lds_b128(
                    to_glb(wlo + g), to_lds(As_lo[j] + a_co * LSTR + a_cg), 0, 0);
#else
                *(uint4*)(As_hi[j] + a_co * LSTR + a_cg) = *(const uint4*)(whi + g);
                *(uint4*)(As_lo[j] + a_co * LSTR + a_cg) = *(const uint4*)(wlo + g);
#endif
            }

            // ---- stage B panel: rows p=1..64 (w_in=0..63), 32 cin wide ----
            {
                union { unsigned short s[8]; uint4 q; } hb, lb;
                if (rv) {                                  // uniform branch
                    const long long base = rowoff + (long long)(cib + b_cg) * HW + b_w;
                    #pragma unroll
                    for (int j = 0; j < 8; ++j) {
                        const float x = in[base + (long long)j * HW];
                        const __bf16 hv = (__bf16)x;
                        hb.s[j] = bf_bits(hv);
                        lb.s[j] = bf_bits((__bf16)(x - (float)hv));
                    }
                } else {
                    hb.q = make_uint4(0, 0, 0, 0);
                    lb.q = make_uint4(0, 0, 0, 0);
                }
                *(uint4*)(Bs_hi + (b_w + 1) * LSTR + b_cg) = hb.q;
                *(uint4*)(Bs_lo + (b_w + 1) * LSTR + b_cg) = lb.q;
            }

#ifdef USE_ASYNC_A
            __builtin_amdgcn_s_wait_asynccnt(0);   // A copies landed in LDS
#endif
            __syncthreads();

            // ---- three dw taps from one panel: 18 WMMAs ----
            #pragma unroll
            for (int j = 0; j < 3; ++j) {
                FragU a_hi, a_lo, b0h, b0l, b1h, b1l;
                const unsigned short* ap = As_hi[j] + (co_sub + m) * LSTR + kb;
                a_hi.q[0] = *(const uint4*)(ap);
                a_hi.q[1] = *(const uint4*)(ap + 16);
                const unsigned short* al = As_lo[j] + (co_sub + m) * LSTR + kb;
                a_lo.q[0] = *(const uint4*)(al);
                a_lo.q[1] = *(const uint4*)(al + 16);

                // B rows: padded index p = w_out + j
                const unsigned short* p0h = Bs_hi + (w_sub + m + j) * LSTR + kb;
                b0h.q[0] = *(const uint4*)(p0h);
                b0h.q[1] = *(const uint4*)(p0h + 16);
                const unsigned short* p0l = Bs_lo + (w_sub + m + j) * LSTR + kb;
                b0l.q[0] = *(const uint4*)(p0l);
                b0l.q[1] = *(const uint4*)(p0l + 16);
                const unsigned short* p1h = Bs_hi + (w_sub + 16 + m + j) * LSTR + kb;
                b1h.q[0] = *(const uint4*)(p1h);
                b1h.q[1] = *(const uint4*)(p1h + 16);
                const unsigned short* p1l = Bs_lo + (w_sub + 16 + m + j) * LSTR + kb;
                b1l.q[0] = *(const uint4*)(p1l);
                b1l.q[1] = *(const uint4*)(p1l + 16);

                // 2-term split product: hi*hi + hi*lo + lo*hi
                d0 = __builtin_amdgcn_wmma_f32_16x16x32_bf16(false, a_hi.v, false, b0h.v, (short)0, d0, false, false);
                d0 = __builtin_amdgcn_wmma_f32_16x16x32_bf16(false, a_hi.v, false, b0l.v, (short)0, d0, false, false);
                d0 = __builtin_amdgcn_wmma_f32_16x16x32_bf16(false, a_lo.v, false, b0h.v, (short)0, d0, false, false);
                d1 = __builtin_amdgcn_wmma_f32_16x16x32_bf16(false, a_hi.v, false, b1h.v, (short)0, d1, false, false);
                d1 = __builtin_amdgcn_wmma_f32_16x16x32_bf16(false, a_hi.v, false, b1l.v, (short)0, d1, false, false);
                d1 = __builtin_amdgcn_wmma_f32_16x16x32_bf16(false, a_lo.v, false, b1h.v, (short)0, d1, false, false);
            }
        }
    }

    // ---- epilogue: bias + coalesced f32 stores (lanes 0-15 contiguous w) ---
    #pragma unroll
    for (int r = 0; r < 8; ++r) {
        const int co = cb + co_sub + r + ((lane >= 16) ? 8 : 0);
        const float bv = bias[co];
        const size_t ob = (((size_t)n * COUT + co) * Hn + h) * (size_t)Wn;
        out[ob + w_sub + m]      = d0[r] + bv;
        out[ob + w_sub + 16 + m] = d1[r] + bv;
    }
}

// ---------------------------------------------------------------------------
// Launch: inputs are {input, conv_w, conv_b, weight} (all fp32).
// Workspace: scale f32[65536] | whi u16[9*65536] | wlo u16[9*65536] (~2.5 MB)
// ---------------------------------------------------------------------------
extern "C" void kernel_launch(void* const* d_in, const int* in_sizes, int n_in,
                              void* d_out, int out_size, void* d_ws, size_t ws_size,
                              hipStream_t stream) {
    const float* input  = (const float*)d_in[0];
    const float* conv_w = (const float*)d_in[1];
    const float* conv_b = (const float*)d_in[2];
    const float* gate   = (const float*)d_in[3];
    float* out = (float*)d_out;

    char* ws = (char*)d_ws;
    float*          scale = (float*)ws;
    unsigned short* whi   = (unsigned short*)(ws + (size_t)COUT * CIN * sizeof(float));
    unsigned short* wlo   = whi + (size_t)9 * COUT * CIN;

    scale_kernel<<<1, 256, 0, stream>>>(gate, scale);
    wpack_kernel<<<COUT, 256, 0, stream>>>(conv_w, scale, whi, wlo);

    // 4 cout tiles x (B*H = 1024) rows = 4096 blocks, 256 threads (8 waves)
    conv_kernel<<<4 * Bn * Hn, 256, 0, stream>>>(input, whi, wlo, conv_b, out);
}
